// MHCLayer_13950053777922
// MI455X (gfx1250) — compile-verified, matrix operand order checked by reading, not run on hardware
//
#include <hip/hip_runtime.h>
#include <hip/hip_bf16.h>

// ---- sizes ----
#define DD     2048
#define NAG    64
#define BB     64
#define ROWS   4096          // BB*NAG
#define NQKV   6144          // 3*DD
static_assert(ROWS == BB * NAG, "");

// ---- WMMA vector types (probe-verified signatures) ----
typedef __attribute__((ext_vector_type(16))) __bf16   v16bf;
typedef __attribute__((ext_vector_type(8)))  float    v8f;
typedef __attribute__((ext_vector_type(4)))  unsigned v4u;

union Frag { v16bf v; v4u q[2]; };

// ============================================================
// Kernel 1: convert wq,wk,wv (fp32 [2048][2048] each) -> bf16 [6144][2048]
// ============================================================
__global__ __launch_bounds__(256) void cvt_weights(
    const float* __restrict__ wq, const float* __restrict__ wk,
    const float* __restrict__ wv, __bf16* __restrict__ W) {
  const size_t n1 = (size_t)DD * DD;
  size_t i = ((size_t)blockIdx.x * 256 + threadIdx.x) * 4;
  const float* src; size_t off;
  if (i < n1)            { src = wq; off = i; }
  else if (i < 2 * n1)   { src = wk; off = i - n1; }
  else                   { src = wv; off = i - 2 * n1; }
  float4 f = *(const float4*)(src + off);
  W[i + 0] = (__bf16)f.x;
  W[i + 1] = (__bf16)f.y;
  W[i + 2] = (__bf16)f.z;
  W[i + 3] = (__bf16)f.w;
}

// ============================================================
// Kernel 2: LayerNorm per row -> fp32 + bf16 copies
// ============================================================
__global__ __launch_bounds__(256) void ln_kernel(
    const float* __restrict__ x, const float* __restrict__ w,
    const float* __restrict__ b, float* __restrict__ nf,
    __bf16* __restrict__ nb) {
  __shared__ float red[256];
  const int tid = threadIdx.x;
  const size_t row = blockIdx.x;
  const float* xr = x + row * DD;

  float v[8]; float s = 0.f;
#pragma unroll
  for (int j = 0; j < 8; ++j) { v[j] = xr[tid + j * 256]; s += v[j]; }
  red[tid] = s; __syncthreads();
  for (int st = 128; st > 0; st >>= 1) { if (tid < st) red[tid] += red[tid + st]; __syncthreads(); }
  const float mean = red[0] * (1.0f / DD); __syncthreads();

  float s2 = 0.f;
#pragma unroll
  for (int j = 0; j < 8; ++j) { float d = v[j] - mean; s2 += d * d; }
  red[tid] = s2; __syncthreads();
  for (int st = 128; st > 0; st >>= 1) { if (tid < st) red[tid] += red[tid + st]; __syncthreads(); }
  const float rstd = rsqrtf(red[0] * (1.0f / DD) + 1e-5f);

#pragma unroll
  for (int j = 0; j < 8; ++j) {
    int d = tid + j * 256;
    float val = (v[j] - mean) * rstd * w[d] + b[d];
    nf[row * DD + d] = val;
    nb[row * DD + d] = (__bf16)val;
  }
}

// ============================================================
// QKV GEMM helpers: 32(M) x 64(N) per wave, K ping-pong double buffer
// ============================================================
__device__ __forceinline__ void load_ab(Frag a[2], Frag b[4],
                                        const __bf16* __restrict__ ar0,
                                        const __bf16* __restrict__ ar1,
                                        const __bf16* const wr[4],
                                        int k0, int kh) {
  a[0].q[0] = *(const v4u*)(ar0 + k0 + kh * 8);
  a[0].q[1] = *(const v4u*)(ar0 + k0 + 16 + kh * 8);
  a[1].q[0] = *(const v4u*)(ar1 + k0 + kh * 8);
  a[1].q[1] = *(const v4u*)(ar1 + k0 + 16 + kh * 8);
#pragma unroll
  for (int t = 0; t < 4; ++t) {
    b[t].q[0] = *(const v4u*)(wr[t] + k0 + kh * 8);
    b[t].q[1] = *(const v4u*)(wr[t] + k0 + 16 + kh * 8);
  }
}

__device__ __forceinline__ void mma_step(v8f acc[2][4], const Frag a[2], const Frag b[4]) {
#pragma unroll
  for (int i = 0; i < 2; ++i)
#pragma unroll
    for (int t = 0; t < 4; ++t)
      acc[i][t] = __builtin_amdgcn_wmma_f32_16x16x32_bf16(
          false, a[i].v, false, b[t].v, (short)0, acc[i][t], false, false);
}

// ============================================================
// Kernel 3: QKV GEMM  C[4096][6144] = A[4096][2048] * W[6144][2048]^T + bias
// One wave per 32(M) x 64(N) tile; 12288 waves = 128 mTiles * 96 nTiles.
// ============================================================
__global__ __launch_bounds__(256) void qkv_gemm(
    const __bf16* __restrict__ A, const __bf16* __restrict__ W,
    const float* __restrict__ bq, const float* __restrict__ bk,
    const float* __restrict__ bv, __bf16* __restrict__ C) {
  const int tid  = threadIdx.x;
  const int lane = tid & 31;
  const int wave = blockIdx.x * 8 + (tid >> 5);
  const int mTile = wave & 127;                   // 128 tiles of 32 rows
  const int nBase = (wave >> 7) * 64;             // 96 tiles of 64 cols
  const int r  = lane & 15;
  const int kh = lane >> 4;

  const __bf16* ar0 = A + (size_t)(mTile * 32 + r) * DD;
  const __bf16* ar1 = A + (size_t)(mTile * 32 + 16 + r) * DD;
  const __bf16* wr[4];
#pragma unroll
  for (int t = 0; t < 4; ++t)
    wr[t] = W + (size_t)(nBase + t * 16 + r) * DD;

  v8f acc[2][4] = {};

  Frag aC[2], bC[4], aN[2], bN[4];
  load_ab(aC, bC, ar0, ar1, wr, 0, kh);
#pragma unroll 1
  for (int k0 = 0; k0 < DD; k0 += 64) {
    load_ab(aN, bN, ar0, ar1, wr, k0 + 32, kh);           // prefetch odd step
    mma_step(acc, aC, bC);
    load_ab(aC, bC, ar0, ar1, wr, (k0 + 64) & (DD - 1), kh); // prefetch next even step
    mma_step(acc, aN, bN);
  }

  const int sel = nBase >> 11;                    // 0:q 1:k 2:v
  const float* bias = (sel == 0) ? bq : (sel == 1) ? bk : bv;
  const int bcol0 = nBase - (sel << 11);
#pragma unroll
  for (int i = 0; i < 2; ++i) {
#pragma unroll
    for (int t = 0; t < 4; ++t) {
      const int col = nBase + t * 16 + (lane & 15);
      const float bb = bias[bcol0 + t * 16 + (lane & 15)];
#pragma unroll
      for (int rr = 0; rr < 8; ++rr) {
        int m = mTile * 32 + i * 16 + rr + 8 * kh;
        C[(size_t)m * NQKV + col] = (__bf16)(acc[i][t][rr] + bb);
      }
    }
  }
}

// ============================================================
// Kernel 4: scores[b][n][m] = q[b,n,:].k[b,m,:] / sqrt(D)
// One wave per (batch, 16x16 tile): 64*16 = 1024 waves; K double-buffered.
// ============================================================
__global__ __launch_bounds__(256) void scores_gemm(
    const __bf16* __restrict__ QKV, float* __restrict__ S) {
  const int tid  = threadIdx.x;
  const int lane = tid & 31;
  const int wave = blockIdx.x * 8 + (tid >> 5);
  const int b     = wave >> 4;
  const int nTile = (wave >> 2) & 3;
  const int mTile = wave & 3;
  const int r  = lane & 15;
  const int kh = lane >> 4;

  const __bf16* qrow = QKV + (size_t)(b * NAG + nTile * 16 + r) * NQKV;        // Q cols 0..2047
  const __bf16* krow = QKV + (size_t)(b * NAG + mTile * 16 + r) * NQKV + DD;   // K cols 2048..4095

  v8f acc = {};
  Frag aC, bC, aN, bN;
  aC.q[0] = *(const v4u*)(qrow + kh * 8);
  aC.q[1] = *(const v4u*)(qrow + 16 + kh * 8);
  bC.q[0] = *(const v4u*)(krow + kh * 8);
  bC.q[1] = *(const v4u*)(krow + 16 + kh * 8);
#pragma unroll 1
  for (int k0 = 0; k0 < DD; k0 += 64) {
    int k1 = k0 + 32;
    aN.q[0] = *(const v4u*)(qrow + k1 + kh * 8);
    aN.q[1] = *(const v4u*)(qrow + k1 + 16 + kh * 8);
    bN.q[0] = *(const v4u*)(krow + k1 + kh * 8);
    bN.q[1] = *(const v4u*)(krow + k1 + 16 + kh * 8);
    acc = __builtin_amdgcn_wmma_f32_16x16x32_bf16(
        false, aC.v, false, bC.v, (short)0, acc, false, false);
    int k2 = (k0 + 64) & (DD - 1);
    aC.q[0] = *(const v4u*)(qrow + k2 + kh * 8);
    aC.q[1] = *(const v4u*)(qrow + k2 + 16 + kh * 8);
    bC.q[0] = *(const v4u*)(krow + k2 + kh * 8);
    bC.q[1] = *(const v4u*)(krow + k2 + 16 + kh * 8);
    acc = __builtin_amdgcn_wmma_f32_16x16x32_bf16(
        false, aN.v, false, bN.v, (short)0, acc, false, false);
  }
  const float rs = 0.022097086912079608f;  // 1/sqrt(2048)
#pragma unroll
  for (int rr = 0; rr < 8; ++rr) {
    int n = nTile * 16 + rr + 8 * kh;
    int m = mTile * 16 + (lane & 15);
    S[(size_t)b * 4096 + (size_t)n * 64 + m] = acc[rr] * rs;
  }
}

// ============================================================
// Kernel 5: Sinkhorn (50 iters, stable logsumexp) -> attn (written to d_out tail)
// ============================================================
__global__ __launch_bounds__(256) void sinkhorn(
    const float* __restrict__ S, float* __restrict__ attn) {
  __shared__ float la[64][65];
  __shared__ float lse[64];
  const int b = blockIdx.x;
  const int tid = threadIdx.x;

  for (int i = tid; i < 4096; i += 256)
    la[i >> 6][i & 63] = S[(size_t)b * 4096 + i];
  __syncthreads();

  for (int it = 0; it < 50; ++it) {
    if (tid < 64) {  // row logsumexp (over m)
      float mx = -3.402823466e38f;
      for (int j = 0; j < 64; ++j) mx = fmaxf(mx, la[tid][j]);
      float s = 0.f;
      for (int j = 0; j < 64; ++j) s += expf(la[tid][j] - mx);
      lse[tid] = mx + logf(s);
    }
    __syncthreads();
    for (int i = tid; i < 4096; i += 256) la[i >> 6][i & 63] -= lse[i >> 6];
    __syncthreads();
    if (tid < 64) {  // col logsumexp (over n)
      float mx = -3.402823466e38f;
      for (int j = 0; j < 64; ++j) mx = fmaxf(mx, la[j][tid]);
      float s = 0.f;
      for (int j = 0; j < 64; ++j) s += expf(la[j][tid] - mx);
      lse[tid] = mx + logf(s);
    }
    __syncthreads();
    for (int i = tid; i < 4096; i += 256) la[i >> 6][i & 63] -= lse[i & 63];
    __syncthreads();
  }

  for (int i = tid; i < 4096; i += 256)
    attn[(size_t)b * 4096 + i] = expf(la[i >> 6][i & 63]);
}

// ============================================================
// Kernel 6: attended = attn @ v ; mix, bias, scale, norm-clamp, residual
// ============================================================
__global__ __launch_bounds__(256) void output_kernel(
    const float* __restrict__ attn, const __bf16* __restrict__ QKV,
    const float* __restrict__ normed, const float* __restrict__ x,
    const float* __restrict__ agent_biases, const float* __restrict__ scale_factors,
    float* __restrict__ out) {
  __shared__ float w[64];
  __shared__ float red[256];
  const int tid = threadIdx.x;
  const int row = blockIdx.x;        // b*64 + n
  const int b = row >> 6;
  const int n = row & 63;

  if (tid < 64) w[tid] = attn[(size_t)row * 64 + tid];
  __syncthreads();

  float sc[8];
#pragma unroll
  for (int j = 0; j < 8; ++j) sc[j] = 0.f;

  for (int m = 0; m < 64; ++m) {
    const float wm = w[m];
    const __bf16* vrow = QKV + (size_t)(b * NAG + m) * NQKV + 2 * DD;
#pragma unroll
    for (int j = 0; j < 8; ++j)
      sc[j] += wm * (float)vrow[tid + j * 256];
  }

  const float sf = scale_factors[n];
  float ss = 0.f;
#pragma unroll
  for (int j = 0; j < 8; ++j) {
    int d = tid + j * 256;
    float mixed  = 0.1f * normed[(size_t)row * DD + d] + 0.9f * sc[j];
    float scaled = (mixed + agent_biases[(size_t)n * DD + d]) * sf;
    sc[j] = scaled;
    ss += scaled * scaled;
  }
  red[tid] = ss; __syncthreads();
  for (int st = 128; st > 0; st >>= 1) { if (tid < st) red[tid] += red[tid + st]; __syncthreads(); }
  const float inv = 1.0f / fmaxf(1.0f, sqrtf(red[0]));   // SIGNAL_BOUND = 1

#pragma unroll
  for (int j = 0; j < 8; ++j) {
    int d = tid + j * 256;
    out[(size_t)row * DD + d] = x[(size_t)row * DD + d] + sc[j] * inv;
  }
}

// ============================================================
// Launcher
// ============================================================
extern "C" void kernel_launch(void* const* d_in, const int* in_sizes, int n_in,
                              void* d_out, int out_size, void* d_ws, size_t ws_size,
                              hipStream_t stream) {
  const float* x    = (const float*)d_in[0];   // agent_states [64,64,2048]
  const float* ln_w = (const float*)d_in[1];
  const float* ln_b = (const float*)d_in[2];
  const float* wq   = (const float*)d_in[3];
  const float* bq   = (const float*)d_in[4];
  const float* wk   = (const float*)d_in[5];
  const float* bk   = (const float*)d_in[6];
  const float* wv   = (const float*)d_in[7];
  const float* bv   = (const float*)d_in[8];
  const float* ab   = (const float*)d_in[9];   // agent_biases [64,2048]
  const float* sfac = (const float*)d_in[10];  // scale_factors [64]

  // workspace layout (bytes)
  char* ws = (char*)d_ws;
  float*  normedF = (float*) (ws + 0);                 //  33,554,432 B
  __bf16* normedB = (__bf16*)(ws + 33554432);          //  16,777,216 B
  __bf16* W       = (__bf16*)(ws + 50331648);          //  25,165,824 B
  __bf16* QKV     = (__bf16*)(ws + 75497472);          //  50,331,648 B
  float*  S       = (float*) (ws + 125829120);         //   1,048,576 B

  float* out  = (float*)d_out;
  float* attn = out + (size_t)ROWS * DD;               // tuple output #2

  cvt_weights  <<<12288, 256, 0, stream>>>(wq, wk, wv, W);
  ln_kernel    <<<ROWS,  256, 0, stream>>>(x, ln_w, ln_b, normedF, normedB);
  qkv_gemm     <<<1536,  256, 0, stream>>>(normedB, W, bq, bk, bv, QKV);
  scores_gemm  <<<128,   256, 0, stream>>>(QKV, S);
  sinkhorn     <<<BB,    256, 0, stream>>>(S, attn);
  output_kernel<<<ROWS,  256, 0, stream>>>(attn, QKV, normedF, x, ab, sfac, out);
}
